// HyperbolicGraphDecoder_28114855919640
// MI455X (gfx1250) — matrix operand order repaired and checked by reading, not compiled, not run on hardware
//
#include <hip/hip_runtime.h>

// ---------------------------------------------------------------------------
// HyperbolicGraphDecoder on MI455X (gfx1250): bf16 WMMA GEMMs + fused
// Poincare log/exp map row kernels. All matrix math via
// v_wmma_f32_16x16x32_bf16 (f32 accumulate).
//
// GEMM: LDS-free. B is consumed in an [N][K] (column-major-of-B) layout so
// every B fragment is two contiguous global_load_b128; weights are
// transposed during fp32->bf16 conversion, and GEMM1 writes its output H
// directly transposed (packed v8bf stores: lane's 8 acc values are 8
// consecutive M). 256-thread blocks (8 waves, 4x2), block tile 256x128,
// wave tile 64x64 (16 accumulators), K-step 32, no barriers in the K-loop.
// ---------------------------------------------------------------------------

typedef __bf16 bf16;
typedef __attribute__((ext_vector_type(16))) __bf16 v16bf;
typedef __attribute__((ext_vector_type(8)))  __bf16 v8bf;
typedef __attribute__((ext_vector_type(4)))  __bf16 v4bf;
typedef __attribute__((ext_vector_type(8)))  float  v8f;

#define HEPS 1e-7f
#define MAX_TANH_ARG (1.0f - 1e-5f)

// ---------------------------------------------------------------------------
// fp32 -> bf16 conversion (vectorized, grid-stride). count4 = n/4.
// ---------------------------------------------------------------------------
__global__ void cvt_f32_bf16(const float4* __restrict__ in, bf16* __restrict__ out,
                             long long count4) {
  long long i = (long long)blockIdx.x * blockDim.x + threadIdx.x;
  long long stride = (long long)gridDim.x * blockDim.x;
  for (; i < count4; i += stride) {
    float4 v = in[i];
    v4bf o;
    o.x = (bf16)v.x; o.y = (bf16)v.y; o.z = (bf16)v.z; o.w = (bf16)v.w;
    *(v4bf*)(out + i * 4) = o;
  }
}

// ---------------------------------------------------------------------------
// fp32 [K][N] -> bf16 [N][K] transpose-convert (32x32 LDS tiles).
// grid = (K/32, N/32), 256 threads (32 x 8).
// ---------------------------------------------------------------------------
__global__ __launch_bounds__(256) void cvt_f32_bf16_T(const float* __restrict__ in,
                                                      bf16* __restrict__ out,
                                                      int K, int N) {
  __shared__ float t[32][33];
  const int k0 = blockIdx.x * 32;
  const int n0 = blockIdx.y * 32;
  const int tx = threadIdx.x & 31;
  const int ty = threadIdx.x >> 5;
  #pragma unroll
  for (int i = ty; i < 32; i += 8)
    t[i][tx] = in[(long long)(k0 + i) * N + n0 + tx];
  __syncthreads();
  #pragma unroll
  for (int i = ty; i < 32; i += 8)
    out[(long long)(n0 + i) * K + k0 + tx] = (bf16)t[tx][i];
}

// ---------------------------------------------------------------------------
// Row-norm helper: block-wide sum of squares (256 threads = 8 waves)
// ---------------------------------------------------------------------------
__device__ __forceinline__ float block_sumsq(float s, float* red) {
  #pragma unroll
  for (int off = 16; off > 0; off >>= 1) s += __shfl_down(s, off);
  int lane = threadIdx.x & 31;
  int wid  = threadIdx.x >> 5;
  if (lane == 0) red[wid] = s;
  __syncthreads();
  float tot = 0.0f;
  #pragma unroll
  for (int i = 0; i < 8; ++i) tot += red[i];
  return tot;
}

// ---------------------------------------------------------------------------
// logmap0 (Poincare ball, c=1), fp32 in -> bf16 out. One block per row.
// ---------------------------------------------------------------------------
__global__ __launch_bounds__(256) void logmap0_rows(const float* __restrict__ X,
                                                    bf16* __restrict__ Y, int D) {
  __shared__ float red[8];
  long long row = blockIdx.x;
  const float* x = X + row * D;
  bf16* y = Y + row * D;
  float s = 0.0f;
  for (int i = threadIdx.x; i < D; i += blockDim.x) { float v = x[i]; s += v * v; }
  float tot = block_sumsq(s, red);
  float n = fmaxf(sqrtf(tot), HEPS);
  float scn = fminf(n, MAX_TANH_ARG);
  float f = atanhf(scn) / n;
  for (int i = threadIdx.x; i < D; i += blockDim.x) y[i] = (bf16)(x[i] * f);
}

// ---------------------------------------------------------------------------
// Fused post-aggregation: relu -> expmap0 -> (next layer's) logmap0.
// ---------------------------------------------------------------------------
__global__ __launch_bounds__(256) void relu_exp_log_rows(const bf16* __restrict__ Hin,
                                                         bf16* __restrict__ Xout, int D) {
  __shared__ float red[8];
  long long row = blockIdx.x;
  const bf16* h = Hin + row * D;
  bf16* y = Xout + row * D;
  float s = 0.0f;
  for (int i = threadIdx.x; i < D; i += blockDim.x) {
    float v = fmaxf((float)h[i], 0.0f);
    s += v * v;
  }
  float tot = block_sumsq(s, red);
  float norm = sqrtf(tot);
  float nc = fmaxf(norm, HEPS);
  float f1 = tanhf(nc) / nc;                 // expmap0 factor
  float n2 = fmaxf(f1 * norm, HEPS);         // norm after expmap0
  float scn = fminf(n2, MAX_TANH_ARG);
  float f2 = atanhf(scn) / n2;               // logmap0 factor
  float f = f1 * f2;
  for (int i = threadIdx.x; i < D; i += blockDim.x)
    y[i] = (bf16)(fmaxf((float)h[i], 0.0f) * f);
}

// ---------------------------------------------------------------------------
// bf16 WMMA GEMM (LDS-free):
//   C = A (MxK, row-major) * B, where B is given as Bt[n][k] ([N][ldb]).
//   Batched: A += b*sA; B k-offset += b*sBk; C += b*sC.
// C_TRANS=false: C row-major [M][N].  C_TRANS=true: C = Ct[n][Mtot*? ]
//   transposed [N][M] with packed 16B stores (used for H -> HT).
// ---------------------------------------------------------------------------
template <bool OUT_F32, bool C_TRANS>
__global__ __launch_bounds__(256) void gemm_bf16_wmma(
    const bf16* __restrict__ A, const bf16* __restrict__ Bt,
    const float* __restrict__ bias, const float* __restrict__ mask,
    void* __restrict__ C, int M, int N, int K,
    long long sA, long long sBk, long long sC, long long ldb) {
  const int b = blockIdx.z;
  A += (long long)b * sA;
  const long long kOff = (long long)b * sBk;

  const int n_blk = blockIdx.x * 128;
  const int m_blk = blockIdx.y * 256;
  const int lane = threadIdx.x & 31;
  const int wid  = threadIdx.x >> 5;
  const int ln   = lane & 15;   // fragment N index / A row select
  const int h    = lane >> 4;   // half-wave per ISA fragment layout
  const int wr   = wid >> 1;    // wave row: m offset = wr*64
  const int wc   = wid & 1;     // wave col: n offset = wc*64

  v8f acc[4][4];
  const v8f vzero = {0.f, 0.f, 0.f, 0.f, 0.f, 0.f, 0.f, 0.f};
  #pragma unroll
  for (int mf = 0; mf < 4; ++mf)
    #pragma unroll
    for (int nf = 0; nf < 4; ++nf) acc[mf][nf] = vzero;

  for (int k0 = 0; k0 < K; k0 += 32) {
    // A fragments: lane (ln,h) holds K = {k0+h*8..+7} and {k0+16+h*8..+7}
    union FR { v16bf v; v8bf p[2]; };
    FR afrag[4];
    #pragma unroll
    for (int mf = 0; mf < 4; ++mf) {
      const int row = m_blk + wr * 64 + mf * 16 + ln;
      const bf16* ap = A + (long long)row * K + k0 + h * 8;
      afrag[mf].p[0] = *(const v8bf*)(ap);
      afrag[mf].p[1] = *(const v8bf*)(ap + 16);
    }
    #pragma unroll
    for (int nf = 0; nf < 4; ++nf) {
      // B fragment: lane (ln,h) holds column n, K = {h*8..+7} and {16+h*8..+7}
      // Bt[n][k] is contiguous in k -> two 16B loads.
      const int n = n_blk + wc * 64 + nf * 16 + ln;
      const bf16* bp = Bt + (long long)n * ldb + kOff + k0 + h * 8;
      FR bfrag;
      bfrag.p[0] = *(const v8bf*)(bp);
      bfrag.p[1] = *(const v8bf*)(bp + 16);
      #pragma unroll
      for (int mf = 0; mf < 4; ++mf) {
        acc[mf][nf] = __builtin_amdgcn_wmma_f32_16x16x32_bf16(
            false, afrag[mf].v, false, bfrag.v,
            (short)0, acc[mf][nf], false, false);
      }
    }
  }

  // Epilogue: C fragment layout -> VGPR r holds M = r + 8*h, lane ln = N
  #pragma unroll
  for (int mf = 0; mf < 4; ++mf) {
    #pragma unroll
    for (int nf = 0; nf < 4; ++nf) {
      const int n = n_blk + wc * 64 + nf * 16 + ln;
      const float bv = bias ? bias[n] : 0.0f;
      if (C_TRANS) {
        // 8 consecutive m per lane -> one packed 16B store into Ct[n][m]
        const int m0 = m_blk + wr * 64 + mf * 16 + 8 * h;
        v8bf pk;
        #pragma unroll
        for (int r = 0; r < 8; ++r) pk[r] = (bf16)(acc[mf][nf][r] + bv);
        *(v8bf*)((bf16*)C + (long long)n * M + m0) = pk;
      } else {
        #pragma unroll
        for (int r = 0; r < 8; ++r) {
          const int m = m_blk + wr * 64 + mf * 16 + 8 * h + r;
          float v = acc[mf][nf][r] + bv;
          const long long idx = (long long)b * sC + (long long)m * N + n;
          if (OUT_F32) {
            if (mask) v *= mask[m];
            ((float*)C)[idx] = v;
          } else {
            ((bf16*)C)[idx] = (bf16)v;
          }
        }
      }
    }
  }
}

// ---------------------------------------------------------------------------
// Launch
// ---------------------------------------------------------------------------
extern "C" void kernel_launch(void* const* d_in, const int* in_sizes, int n_in,
                              void* d_out, int out_size, void* d_ws, size_t ws_size,
                              hipStream_t stream) {
  (void)in_sizes; (void)n_in; (void)out_size; (void)ws_size;

  const float* latent = (const float*)d_in[0];   // [32,1024,256]
  const float* adj    = (const float*)d_in[1];   // [32,1024,1024]
  const float* mask   = (const float*)d_in[2];   // [32,1024,1]
  const float* W[3]   = {(const float*)d_in[3], (const float*)d_in[5], (const float*)d_in[7]};
  const float* bI[3]  = {(const float*)d_in[4], (const float*)d_in[6], (const float*)d_in[8]};
  const float* Wo     = (const float*)d_in[9];   // [512,128]
  const float* bo     = (const float*)d_in[10];  // [128]
  float* out = (float*)d_out;                    // [32,1024,128]

  const int Bz = 32, Nn = 1024, DH = 512, DO = 128;
  const long long ROWS = (long long)Bz * Nn;     // 32768

  // workspace layout (bytes)
  char* ws = (char*)d_ws;
  bf16* adjB = (bf16*)(ws);                                          // 64 MB
  bf16* XB   = (bf16*)(ws + 67108864);                               // 32 MB  [32768][<=512] row-major
  bf16* HT   = (bf16*)(ws + 67108864 + 33554432);                    // 32 MB  [512][32768] transposed
  bf16* YB   = (bf16*)(ws + 67108864 + 2LL * 33554432);              // 32 MB  [32768][512] row-major
  bf16* WT   = (bf16*)(ws + 67108864 + 3LL * 33554432);              // 0.5 MB [N][K] transposed weights

  // adjacency -> bf16 (every call; no cross-call state)
  cvt_f32_bf16<<<4096, 256, 0, stream>>>((const float4*)adj, adjB,
                                         (long long)Bz * Nn * Nn / 4);
  // logmap0 of latent points -> bf16 activations
  logmap0_rows<<<(int)ROWS, 256, 0, stream>>>(latent, XB, 256);

  int Din = 256;
  for (int l = 0; l < 3; ++l) {
    // W [Din][512] -> WT [512][Din] bf16
    cvt_f32_bf16_T<<<dim3(Din / 32, DH / 32), 256, 0, stream>>>(W[l], WT, Din, DH);
    // HT = (X * W + b)^T : stored [512][32768]
    gemm_bf16_wmma<false, true><<<dim3(DH / 128, (int)(ROWS / 256), 1), 256, 0, stream>>>(
        XB, WT, bI[l], nullptr, HT, (int)ROWS, DH, Din, 0, 0, 0, Din);
    // Y[b] = adj[b] * H[b] : batched [1024 x 512]; B fragments read from HT rows
    gemm_bf16_wmma<false, false><<<dim3(DH / 128, Nn / 256, Bz), 256, 0, stream>>>(
        adjB, HT, nullptr, nullptr, YB, Nn, DH, Nn,
        (long long)Nn * Nn, (long long)Nn, (long long)Nn * DH, ROWS);
    // relu -> expmap0 -> (next) logmap0, fused
    relu_exp_log_rows<<<(int)ROWS, 256, 0, stream>>>(YB, XB, DH);
    Din = DH;
  }

  // output projection: out = X * Wo + bo, masked, fp32
  cvt_f32_bf16_T<<<dim3(DH / 32, DO / 32), 256, 0, stream>>>(Wo, WT, DH, DO);
  gemm_bf16_wmma<true, false><<<dim3(DO / 128, (int)(ROWS / 256), 1), 256, 0, stream>>>(
      XB, WT, bo, mask, out, (int)ROWS, DO, DH, 0, 0, 0, DH);
}
